// ActorNetwork_37804302139538
// MI455X (gfx1250) — matrix operand order repaired and checked by reading, not compiled, not run on hardware
//
#include <hip/hip_runtime.h>
#include <math.h>

typedef __attribute__((ext_vector_type(16))) __bf16 v16bf;
typedef __attribute__((ext_vector_type(8)))  __bf16 v8bf;
typedef __attribute__((ext_vector_type(4)))  __bf16 v4bf;
typedef __attribute__((ext_vector_type(8)))  float  v8f;

// ---------------- degree / symmetric norm ----------------
__global__ void deg_init_kernel(float* deg, int N) {
    int i = blockIdx.x * blockDim.x + threadIdx.x;
    if (i < N) deg[i] = 1.0f;  // self-loop contributes 1
}

__global__ void deg_edge_kernel(const long long* __restrict__ dst, float* deg, long long E) {
    long long i = (long long)blockIdx.x * blockDim.x + threadIdx.x;
    if (i < E) atomicAdd(&deg[(int)dst[i]], 1.0f);
}

__global__ void rsqrt_kernel(float* deg, int N) {
    int i = blockIdx.x * blockDim.x + threadIdx.x;
    if (i < N) deg[i] = rsqrtf(deg[i]);
}

// ---------------- weight packing: W[K,N] f32 -> Wt[N][K] bf16 (B-fragment friendly) -------
__global__ void pack_w_kernel(const float* __restrict__ W, __bf16* __restrict__ Wt,
                              int K, int Ncols) {
    int i = blockIdx.x * blockDim.x + threadIdx.x;
    if (i < K * Ncols) {
        int n = i / K, k = i % K;
        Wt[i] = (__bf16)W[(size_t)k * Ncols + n];
    }
}

// ---------------- WMMA GEMM, f32 A (layer 1: A = x, read-once, keep f32) ------------------
template<int KDIM, int NDIM, bool BIAS, bool RELU>
__global__ void wmma_gemm_f32a_kernel(const float* __restrict__ A,
                                      const __bf16* __restrict__ Wt,  // [NDIM][KDIM] bf16
                                      const float* __restrict__ bias,
                                      float* __restrict__ Out,
                                      int rows) {
    constexpr int COL_TILES = NDIM / 16;
    const int wavesPerBlock = blockDim.x >> 5;
    const int tile = blockIdx.x * wavesPerBlock + (threadIdx.x >> 5);
    const int totalTiles = (rows >> 4) * COL_TILES;
    if (tile >= totalTiles) return;             // wave-uniform: EXEC all-ones for WMMA

    const int rowTile = tile / COL_TILES;
    const int colTile = tile % COL_TILES;
    const int lane = threadIdx.x & 31;
    const int g  = lane >> 4;
    const int ln = lane & 15;
    const int r0 = rowTile << 4;

    const float*  arow = A + (size_t)(r0 + ln) * KDIM + 16 * g;
    const __bf16* wrow = Wt + (size_t)(colTile * 16 + ln) * KDIM + 16 * g;
    v8f acc = {};

    #pragma unroll
    for (int kb = 0; kb < KDIM; kb += 32) {
        v16bf a;
        #pragma unroll
        for (int i = 0; i < 8; ++i)  a[i] = (__bf16)arow[kb + i];       // K = kb+16g+i
        #pragma unroll
        for (int i = 8; i < 16; ++i) a[i] = (__bf16)arow[kb + i];       // K = kb+16g+i
        v16bf b = *(const v16bf*)(wrow + kb);                           // K = kb+16g+i, contiguous
        acc = __builtin_amdgcn_wmma_f32_16x16x32_bf16(
            false, a, false, b, (short)0, acc, false, false);
    }

    const int n = colTile * 16 + ln;
    const float bv = BIAS ? bias[n] : 0.0f;
    #pragma unroll
    for (int j = 0; j < 8; ++j) {
        float v = acc[j] + bv;
        if (RELU) v = fmaxf(v, 0.0f);
        Out[(size_t)(r0 + 8 * g + j) * NDIM + n] = v;
    }
}

// ---------------- WMMA GEMM, bf16 A (layers 2-4), optional bf16 output --------------------
template<int KDIM, int NDIM, bool BIAS, bool RELU, bool OUTBF>
__global__ void wmma_gemm_bf16a_kernel(const __bf16* __restrict__ A,   // [rows][KDIM] bf16
                                       const __bf16* __restrict__ Wt,  // [NDIM][KDIM] bf16
                                       const float* __restrict__ bias,
                                       float* __restrict__ Outf,
                                       __bf16* __restrict__ Outb,
                                       int rows) {
    constexpr int COL_TILES = NDIM / 16;
    const int wavesPerBlock = blockDim.x >> 5;
    const int tile = blockIdx.x * wavesPerBlock + (threadIdx.x >> 5);
    const int totalTiles = (rows >> 4) * COL_TILES;
    if (tile >= totalTiles) return;

    const int rowTile = tile / COL_TILES;
    const int colTile = tile % COL_TILES;
    const int lane = threadIdx.x & 31;
    const int g  = lane >> 4;
    const int ln = lane & 15;
    const int r0 = rowTile << 4;

    const __bf16* arow = A + (size_t)(r0 + ln) * KDIM + 16 * g;
    const __bf16* wrow = Wt + (size_t)(colTile * 16 + ln) * KDIM + 16 * g;
    v8f acc = {};

    #pragma unroll
    for (int kb = 0; kb < KDIM; kb += 32) {
        v8bf lo = *(const v8bf*)(arow + kb);        // K = kb+16g+0..7
        v8bf hi = *(const v8bf*)(arow + kb + 8);    // K = kb+16g+8..15
        v16bf a;
        #pragma unroll
        for (int i = 0; i < 8; ++i) { a[i] = lo[i]; a[i + 8] = hi[i]; }
        v16bf b = *(const v16bf*)(wrow + kb);
        acc = __builtin_amdgcn_wmma_f32_16x16x32_bf16(
            false, a, false, b, (short)0, acc, false, false);
    }

    const int n = colTile * 16 + ln;
    const float bv = BIAS ? bias[n] : 0.0f;
    #pragma unroll
    for (int j = 0; j < 8; ++j) {
        float v = acc[j] + bv;
        if (RELU) v = fmaxf(v, 0.0f);
        if (OUTBF) Outb[(size_t)(r0 + 8 * g + j) * NDIM + n] = (__bf16)v;
        else       Outf[(size_t)(r0 + 8 * g + j) * NDIM + n] = v;
    }
}

// ---------------- aggregation ------------------------------------------------------------
// init: out = bias + h * dinv^2  (self-loop term), float4 vectorized
template<int F>
__global__ void agg_init_kernel(const float* __restrict__ h, const float* __restrict__ dinv,
                                const float* __restrict__ bias, float* __restrict__ out, int N) {
    long long q = (long long)blockIdx.x * blockDim.x + threadIdx.x;   // quad index
    if (q >= (long long)N * (F / 4)) return;
    int n  = (int)(q / (F / 4));
    int f4 = (int)(q % (F / 4)) * 4;
    float di = dinv[n];
    float d2 = di * di;
    float4 hv = ((const float4*)h)[q];
    float4 bv = *(const float4*)(bias + f4);
    float4 o;
    o.x = bv.x + hv.x * d2;
    o.y = bv.y + hv.y * d2;
    o.z = bv.z + hv.z * d2;
    o.w = bv.w + hv.w * d2;
    ((float4*)out)[q] = o;
}

// one wave32 per edge: scalar edge indices + norm, lanes cover features
template<int F>
__global__ void agg_edge_kernel(const long long* __restrict__ src, const long long* __restrict__ dst,
                                const float* __restrict__ h, const float* __restrict__ dinv,
                                float* __restrict__ out, long long E) {
    const int wavesPerBlock = blockDim.x >> 5;
    long long wid = (long long)blockIdx.x * wavesPerBlock + (threadIdx.x >> 5);
    if (wid >= E) return;                        // wave-uniform
    int lane = threadIdx.x & 31;
    int s = __builtin_amdgcn_readfirstlane((int)src[wid]);
    int d = __builtin_amdgcn_readfirstlane((int)dst[wid]);
    float nrm = dinv[s] * dinv[d];               // scalar
    const float* hs = h + (size_t)s * F;
    float* od = out + (size_t)d * F;
    if (F == 32) {
        atomicAdd(&od[lane], hs[lane] * nrm);
    } else {
        float2 hv = ((const float2*)hs)[lane];
        atomicAdd(&od[2 * lane],     hv.x * nrm);
        atomicAdd(&od[2 * lane + 1], hv.y * nrm);
    }
}

// relu + convert to packed bf16 (feeds next WMMA GEMM's A), float4 -> 4x bf16
__global__ void relu_to_bf16_kernel(const float* __restrict__ in, __bf16* __restrict__ out,
                                    long long nq) {   // nq = n/4
    long long i = (long long)blockIdx.x * blockDim.x + threadIdx.x;
    if (i >= nq) return;
    float4 v = ((const float4*)in)[i];
    v4bf o;
    o[0] = (__bf16)fmaxf(v.x, 0.0f);
    o[1] = (__bf16)fmaxf(v.y, 0.0f);
    o[2] = (__bf16)fmaxf(v.z, 0.0f);
    o[3] = (__bf16)fmaxf(v.w, 0.0f);
    ((v4bf*)out)[i] = o;
}

// ---------------- softmax over the flat 6.4M-logit vector --------------------------------
__global__ void reduce_max_partial(const float* __restrict__ x, long long n, float* part) {
    __shared__ float sm[256];
    float m = -3.402823466e38f;
    for (long long i = (long long)blockIdx.x * blockDim.x + threadIdx.x; i < n;
         i += (long long)gridDim.x * blockDim.x)
        m = fmaxf(m, x[i]);
    sm[threadIdx.x] = m; __syncthreads();
    for (int s = 128; s; s >>= 1) {
        if ((int)threadIdx.x < s) sm[threadIdx.x] = fmaxf(sm[threadIdx.x], sm[threadIdx.x + s]);
        __syncthreads();
    }
    if (threadIdx.x == 0) part[blockIdx.x] = sm[0];
}

__global__ void reduce_max_final(const float* __restrict__ part, int nb, float* stats) {
    __shared__ float sm[256];
    float m = -3.402823466e38f;
    for (int i = threadIdx.x; i < nb; i += 256) m = fmaxf(m, part[i]);
    sm[threadIdx.x] = m; __syncthreads();
    for (int s = 128; s; s >>= 1) {
        if ((int)threadIdx.x < s) sm[threadIdx.x] = fmaxf(sm[threadIdx.x], sm[threadIdx.x + s]);
        __syncthreads();
    }
    if (threadIdx.x == 0) stats[0] = sm[0];
}

__global__ void reduce_sumexp_partial(const float* __restrict__ x, long long n,
                                      const float* __restrict__ stats, float* part) {
    __shared__ float sm[256];
    float mx = stats[0];
    float acc = 0.0f;
    for (long long i = (long long)blockIdx.x * blockDim.x + threadIdx.x; i < n;
         i += (long long)gridDim.x * blockDim.x)
        acc += expf(x[i] - mx);
    sm[threadIdx.x] = acc; __syncthreads();
    for (int s = 128; s; s >>= 1) {
        if ((int)threadIdx.x < s) sm[threadIdx.x] += sm[threadIdx.x + s];
        __syncthreads();
    }
    if (threadIdx.x == 0) part[blockIdx.x] = sm[0];
}

__global__ void reduce_sum_final(const float* __restrict__ part, int nb, float* stats) {
    __shared__ float sm[256];
    float acc = 0.0f;
    for (int i = threadIdx.x; i < nb; i += 256) acc += part[i];
    sm[threadIdx.x] = acc; __syncthreads();
    for (int s = 128; s; s >>= 1) {
        if ((int)threadIdx.x < s) sm[threadIdx.x] += sm[threadIdx.x + s];
        __syncthreads();
    }
    if (threadIdx.x == 0) stats[1] = sm[0];
}

__global__ void softmax_final(float* x, long long n, const float* __restrict__ stats) {
    long long i = (long long)blockIdx.x * blockDim.x + threadIdx.x;
    if (i < n) x[i] = expf(x[i] - stats[0]) * (1.0f / stats[1]);
}

// -----------------------------------------------------------------------------------------
static inline int cdiv_ll(long long a, int b) { return (int)((a + b - 1) / b); }

extern "C" void kernel_launch(void* const* d_in, const int* in_sizes, int n_in,
                              void* d_out, int out_size, void* d_ws, size_t ws_size,
                              hipStream_t stream) {
    constexpr int IN = 256, H1 = 32, H2 = 64, ACT = 64;

    const float*     x   = (const float*)d_in[0];
    const long long* ei  = (const long long*)d_in[1];   // int64 [2, E]
    const float*     W1  = (const float*)d_in[3];
    const float*     b1  = (const float*)d_in[4];
    const float*     W2  = (const float*)d_in[5];
    const float*     b2  = (const float*)d_in[6];
    const float*     Wo1 = (const float*)d_in[7];
    const float*     bo1 = (const float*)d_in[8];
    const float*     Wo2 = (const float*)d_in[9];
    const float*     bo2 = (const float*)d_in[10];

    const int       N = in_sizes[0] / IN;               // 100000
    const long long E = (long long)in_sizes[1] / 2;     // 3200000
    const long long* src = ei;
    const long long* dst = ei + E;

    // workspace layout
    char* ws = (char*)d_ws;
    size_t off = 0;
    float* dinv = (float*)(ws + off);  off += (((size_t)N * 4) + 255) & ~(size_t)255;
    float* bufA = (float*)(ws + off);  off += (size_t)N * 64 * 4;         // GEMM f32 outputs
    float* bufB = (float*)(ws + off);  off += (size_t)N * 64 * 4;         // agg accumulators
    __bf16* bufC = (__bf16*)(ws + off); off += (size_t)N * 64 * 2;        // bf16 activations
    __bf16* bufD = (__bf16*)(ws + off); off += (size_t)N * 64 * 2;        // bf16 activations
    __bf16* Wt1  = (__bf16*)(ws + off); off += (size_t)IN * H1 * 2;
    __bf16* Wt2  = (__bf16*)(ws + off); off += (size_t)H1 * H2 * 2;
    __bf16* Wto1 = (__bf16*)(ws + off); off += (size_t)H2 * H2 * 2;
    __bf16* Wto2 = (__bf16*)(ws + off); off += (size_t)H2 * ACT * 2;
    off = (off + 255) & ~(size_t)255;
    float* part  = (float*)(ws + off);                  // 1024 floats
    float* stats = part + 1024;                         // [0]=max, [1]=sum
    (void)ws_size;

    float* logits = (float*)d_out;                      // [N, 64]

    const int blkN = (N + 255) / 256;
    const int blkE = cdiv_ll(E, 256);

    // 0) pack weights to bf16 column-major (tiny)
    pack_w_kernel<<<(IN * H1 + 255) / 256, 256, 0, stream>>>(W1, Wt1, IN, H1);
    pack_w_kernel<<<(H1 * H2 + 255) / 256, 256, 0, stream>>>(W2, Wt2, H1, H2);
    pack_w_kernel<<<(H2 * H2 + 255) / 256, 256, 0, stream>>>(Wo1, Wto1, H2, H2);
    pack_w_kernel<<<(H2 * ACT + 255) / 256, 256, 0, stream>>>(Wo2, Wto2, H2, ACT);

    // 1) degrees -> dinv
    deg_init_kernel<<<blkN, 256, 0, stream>>>(dinv, N);
    deg_edge_kernel<<<blkE, 256, 0, stream>>>(dst, dinv, E);
    rsqrt_kernel<<<blkN, 256, 0, stream>>>(dinv, N);

    const int rowTiles = N / 16;                        // 6250, exact
    const int edgeBlocks = cdiv_ll(E, 8);               // 8 waves (edges) per block

    // 2) h1 = x @ W1  [N,32] -> bufA
    {
        int tiles = rowTiles * (H1 / 16);
        wmma_gemm_f32a_kernel<IN, H1, false, false><<<(tiles + 3) / 4, 128, 0, stream>>>(
            x, Wt1, nullptr, bufA, N);
    }
    // 3) z1 = relu(agg(h1) + b1) -> bufB (f32), bufC (bf16)
    agg_init_kernel<H1><<<cdiv_ll((long long)N * H1 / 4, 256), 256, 0, stream>>>(bufA, dinv, b1, bufB, N);
    agg_edge_kernel<H1><<<edgeBlocks, 256, 0, stream>>>(src, dst, bufA, dinv, bufB, E);
    relu_to_bf16_kernel<<<cdiv_ll((long long)N * H1 / 4, 256), 256, 0, stream>>>(
        bufB, bufC, (long long)N * H1 / 4);

    // 4) h2 = z1 @ W2  [N,64] -> bufA
    {
        int tiles = rowTiles * (H2 / 16);
        wmma_gemm_bf16a_kernel<H1, H2, false, false, false><<<(tiles + 3) / 4, 128, 0, stream>>>(
            bufC, Wt2, nullptr, bufA, nullptr, N);
    }
    // 5) z2 = relu(agg(h2) + b2) -> bufB (f32), bufC (bf16)
    agg_init_kernel<H2><<<cdiv_ll((long long)N * H2 / 4, 256), 256, 0, stream>>>(bufA, dinv, b2, bufB, N);
    agg_edge_kernel<H2><<<edgeBlocks, 256, 0, stream>>>(src, dst, bufA, dinv, bufB, E);
    relu_to_bf16_kernel<<<cdiv_ll((long long)N * H2 / 4, 256), 256, 0, stream>>>(
        bufB, bufC, (long long)N * H2 / 4);

    // 6) t = relu(z2 @ Wo1 + bo1) -> bufD (bf16)
    {
        int tiles = rowTiles * (H2 / 16);
        wmma_gemm_bf16a_kernel<H2, H2, true, true, true><<<(tiles + 3) / 4, 128, 0, stream>>>(
            bufC, Wto1, bo1, nullptr, bufD, N);
    }
    // 7) logits = t @ Wo2 + bo2 -> d_out (f32)
    {
        int tiles = rowTiles * (ACT / 16);
        wmma_gemm_bf16a_kernel<H2, ACT, true, false, false><<<(tiles + 3) / 4, 128, 0, stream>>>(
            bufD, Wto2, bo2, logits, nullptr, N);
    }

    // 8) softmax over all N*64 logits, in place
    const long long M = (long long)N * ACT;
    reduce_max_partial<<<1024, 256, 0, stream>>>(logits, M, part);
    reduce_max_final<<<1, 256, 0, stream>>>(part, 1024, stats);
    reduce_sumexp_partial<<<1024, 256, 0, stream>>>(logits, M, stats, part);
    reduce_sum_final<<<1, 256, 0, stream>>>(part, 1024, stats);
    softmax_final<<<cdiv_ll(M, 256), 256, 0, stream>>>(logits, M, stats);
}
